// LevelAttention_loss_85538568667832
// MI455X (gfx1250) — matrix operand order repaired and checked by reading, not compiled, not run on hardware
//
#include <hip/hip_runtime.h>
#include <hip/hip_bf16.h>
#include <stdint.h>

typedef __attribute__((ext_vector_type(16))) _Float16 v16h;
typedef __attribute__((ext_vector_type(8)))  float    v8f;

#define B_SZ   16
#define N_BOX  96
#define N_LVL  5
#define EPSV   1e-7f

// Workspace layout:
//   uint64_t boxinfo[N_LVL][B_SZ][N_BOX]      61440 B (packed rects; 0 == unselected)
//   float    partial[N_LVL][B_SZ]               320 B
//   int      hasbox[B_SZ]                        64 B
//   (optional, if ws_size allows)
//   A/B WMMA fragments per (level, batch, tile-row/col), pre-laid-out in the
//   ISA 16-bit A16x32 / B32x16 register order: [3 chunks][32 lanes][32 B]

// ---------------------------------------------------------------------------
// Per-(level,batch,box) preprocessing: selection band + integer rect in level
// coordinates. Also zeroes the partial accumulators and computes has_box.
// ---------------------------------------------------------------------------
__global__ void prep_boxes(const float* __restrict__ bboxs,
                           const int* __restrict__ ph, const int* __restrict__ pw,
                           uint64_t* __restrict__ boxinfo,
                           float* __restrict__ partial,
                           int* __restrict__ hasbox)
{
    int lvl = blockIdx.x;      // 0..4
    int b   = blockIdx.y;      // 0..B-1
    int n   = threadIdx.x;     // 0..95
    __shared__ int s_has;
    if (n == 0) {
        partial[lvl * B_SZ + b] = 0.0f;
        s_has = 0;
    }
    __syncthreads();

    float hf = (float)(*ph);
    float wf = (float)(*pw);
    int   side = 1024 >> lvl;
    float sx = (float)side / wf;
    float sy = (float)side / hf;

    const float* bb = bboxs + ((size_t)b * N_BOX + n) * 5;
    float x1 = bb[0], y1 = bb[1], x2 = bb[2], y2 = bb[3], label = bb[4];
    bool valid = (label != -1.0f) && (x1 <= wf) && (y1 <= hf) && (x2 <= wf) && (y2 <= hf);
    float area = (x2 - x1) * (y2 - y1);

    float base = (float)(32 << lvl);
    float min_area = base * base * 0.5f;
    float max_area = (base * 1.58f) * (base * 1.58f) * 2.0f;
    bool sel = valid && (area >= min_area) && (area <= max_area);

    uint64_t pack = 0;   // empty rect => box contributes nothing
    if (sel) {
        int ix1 = (int)fmaxf(floorf(x1 * sx), 0.0f);
        int iy1 = (int)fmaxf(floorf(y1 * sy), 0.0f);
        int ix2 = (int)fminf(ceilf(x2 * sx) + 1.0f, (float)side);
        int iy2 = (int)fminf(ceilf(y2 * sy) + 1.0f, (float)side);
        pack =  (uint64_t)(uint16_t)ix1
             | ((uint64_t)(uint16_t)ix2 << 16)
             | ((uint64_t)(uint16_t)iy1 << 32)
             | ((uint64_t)(uint16_t)iy2 << 48);
    }
    boxinfo[((size_t)lvl * B_SZ + b) * N_BOX + n] = pack;

    if (lvl == 0 && valid) atomicOr(&s_has, 1);
    __syncthreads();
    if (lvl == 0 && n == 0) hasbox[b] = s_has;
}

// ---------------------------------------------------------------------------
// Build WMMA fragments once per (level,batch,tile-row/col). One wave per
// (b, t): lane layout matches the ISA's 16-bit A 16x32 and B 32x16 register
// striping so the main kernel loads them with plain contiguous vector loads.
// ---------------------------------------------------------------------------
__global__ void build_frags(const uint64_t* __restrict__ boxinfo,
                            uint8_t* __restrict__ afrag,
                            uint8_t* __restrict__ bfrag,
                            int side, int lvl)
{
    int t    = blockIdx.x;     // tile-row (for A) == tile-col (for B) index
    int b    = blockIdx.y;
    int lane = threadIdx.x;    // 0..31, one wave
    const uint64_t* gb = boxinfo + ((size_t)lvl * B_SZ + b) * N_BOX;
    int nt = side >> 4;
    int v  = (t << 4) + (lane & 15);       // row y for A / col x for B
    int kbA = (lane < 16) ? 0 : 8;
    int kbB = (lane < 16) ? 0 : 8;

    #pragma unroll
    for (int c = 0; c < 3; ++c) {
        int kc = c * 32;
        v16h a, bv;
        #pragma unroll
        for (int e = 0; e < 16; ++e) {
            uint64_t da = gb[kc + kbA + (e < 8 ? e : e + 8)];
            int ry1 = (int)((da >> 32) & 0xFFFF);
            int ry2 = (int)((da >> 48) & 0xFFFF);
            a[e] = (v >= ry1 && v < ry2) ? (_Float16)1.0f : (_Float16)0.0f;

            uint64_t db = gb[kc + kbB + (e < 8 ? e : e + 8)];
            int rx1 = (int)(db & 0xFFFF);
            int rx2 = (int)((db >> 16) & 0xFFFF);
            bv[e] = (v >= rx1 && v < rx2) ? (_Float16)1.0f : (_Float16)0.0f;
        }
        size_t o = (((size_t)b * nt + t) * 3 + c) * 1024 + (size_t)lane * 32;
        *(v16h*)(afrag + o) = a;
        *(v16h*)(bfrag + o) = bv;
    }
}

// ---------------------------------------------------------------------------
// Main per-level BCE, fragment-load path. All loads (8 att pixels + 6
// fragment vectors) are issued up front so the compiler can clause them and
// hide latency under the 3 chained WMMAs and the log transcendentals.
// ---------------------------------------------------------------------------
__global__ void __launch_bounds__(256)
level_bce_frag(const float* __restrict__ att, int side, int lvl,
               const uint8_t* __restrict__ afrag,
               const uint8_t* __restrict__ bfrag,
               float* __restrict__ partial)
{
    int b    = blockIdx.y;
    int wid  = threadIdx.x >> 5;
    int lane = threadIdx.x & 31;
    int ntx  = side >> 4;
    int nt   = ntx * ntx;
    int tile = blockIdx.x * 8 + wid;     // wave-uniform guard keeps EXEC full
    float acc = 0.0f;

    if (tile < nt) {
        int ty = tile / ntx, tx = tile - ty * ntx;

        // ---- issue attention-map loads first (independent of WMMA) --------
        // C/D layout: lane L -> col = L&15 ; VGPR j -> row = j + (L<16 ? 0 : 8)
        const float* ap = att + (size_t)b * side * side;
        int colx  = (tx << 4) + (lane & 15);
        int rbase = (ty << 4) + ((lane < 16) ? 0 : 8);
        float pv[8];
        #pragma unroll
        for (int j = 0; j < 8; ++j)
            pv[j] = ap[(size_t)(rbase + j) * side + colx];

        // ---- load all 6 fragment vectors, then run 3 back-to-back WMMAs ---
        const uint8_t* ap8 = afrag + (((size_t)b * ntx + ty) * 3) * 1024 + (size_t)lane * 32;
        const uint8_t* bp8 = bfrag + (((size_t)b * ntx + tx) * 3) * 1024 + (size_t)lane * 32;
        v16h a0 = *(const v16h*)(ap8);
        v16h b0 = *(const v16h*)(bp8);
        v16h a1 = *(const v16h*)(ap8 + 1024);
        v16h b1 = *(const v16h*)(bp8 + 1024);
        v16h a2 = *(const v16h*)(ap8 + 2048);
        v16h b2 = *(const v16h*)(bp8 + 2048);

        v8f c = {};
        c = __builtin_amdgcn_wmma_f32_16x16x32_f16(false, a0, false, b0, (short)0, c, false, false);
        c = __builtin_amdgcn_wmma_f32_16x16x32_f16(false, a1, false, b1, (short)0, c, false, false);
        c = __builtin_amdgcn_wmma_f32_16x16x32_f16(false, a2, false, b2, (short)0, c, false, false);

        // ---- BCE on this lane's 8 pixels ----------------------------------
        #pragma unroll
        for (int j = 0; j < 8; ++j) {
            float p = fminf(fmaxf(pv[j], EPSV), 1.0f - EPSV);
            float m = (c[j] > 0.0f) ? 1.0f : 0.0f;
            acc -= m * logf(p) + (1.0f - m) * log1pf(-p);
        }
    }

    #pragma unroll
    for (int off = 16; off > 0; off >>= 1)
        acc += __shfl_xor(acc, off, 32);
    if (lane == 0 && acc != 0.0f)
        atomicAdd(&partial[lvl * B_SZ + b], acc);
}

// ---------------------------------------------------------------------------
// Fallback path (small workspace): build fragments in-tile from LDS box data.
// ---------------------------------------------------------------------------
__global__ void __launch_bounds__(256)
level_bce_inplace(const float* __restrict__ att, int side, int lvl,
                  const uint64_t* __restrict__ boxinfo,
                  float* __restrict__ partial)
{
    int b = blockIdx.y;
    __shared__ uint64_t sbox[N_BOX];
    const uint64_t* gb = boxinfo + ((size_t)lvl * B_SZ + b) * N_BOX;
    if (threadIdx.x < N_BOX) sbox[threadIdx.x] = gb[threadIdx.x];
    __syncthreads();

    int wid  = threadIdx.x >> 5;
    int lane = threadIdx.x & 31;
    int ntx  = side >> 4;
    int nt   = ntx * ntx;
    int tile = blockIdx.x * 8 + wid;
    float acc = 0.0f;

    if (tile < nt) {
        int ty = tile / ntx, tx = tile - ty * ntx;
        int y0 = ty << 4, x0 = tx << 4;
        int mrow = lane & 15;
        int y = y0 + mrow, x = x0 + mrow;
        int kbA = (lane < 16) ? 0 : 8;
        int kbB = (lane < 16) ? 0 : 8;

        v8f c = {};
        #pragma unroll
        for (int kc = 0; kc < N_BOX; kc += 32) {
            v16h a, bv;
            #pragma unroll
            for (int e = 0; e < 16; ++e) {
                uint64_t da = sbox[kc + kbA + (e < 8 ? e : e + 8)];
                int ry1 = (int)((da >> 32) & 0xFFFF);
                int ry2 = (int)((da >> 48) & 0xFFFF);
                a[e] = (y >= ry1 && y < ry2) ? (_Float16)1.0f : (_Float16)0.0f;
                uint64_t db = sbox[kc + kbB + (e < 8 ? e : e + 8)];
                int rx1 = (int)(db & 0xFFFF);
                int rx2 = (int)((db >> 16) & 0xFFFF);
                bv[e] = (x >= rx1 && x < rx2) ? (_Float16)1.0f : (_Float16)0.0f;
            }
            c = __builtin_amdgcn_wmma_f32_16x16x32_f16(
                    false, a, false, bv, (short)0, c, false, false);
        }

        const float* ap = att + (size_t)b * side * side;
        int colx  = x0 + (lane & 15);
        int rbase = y0 + ((lane < 16) ? 0 : 8);
        #pragma unroll
        for (int j = 0; j < 8; ++j) {
            float p = ap[(size_t)(rbase + j) * side + colx];
            p = fminf(fmaxf(p, EPSV), 1.0f - EPSV);
            float m = (c[j] > 0.0f) ? 1.0f : 0.0f;
            acc -= m * logf(p) + (1.0f - m) * log1pf(-p);
        }
    }

    #pragma unroll
    for (int off = 16; off > 0; off >>= 1)
        acc += __shfl_xor(acc, off, 32);
    if (lane == 0 && acc != 0.0f)
        atomicAdd(&partial[lvl * B_SZ + b], acc);
}

// ---------------------------------------------------------------------------
// Combine: per-sample mean over pixels and levels, has_box gating, batch mean.
// ---------------------------------------------------------------------------
__global__ void finalize_loss(const float* __restrict__ partial,
                              const int* __restrict__ hasbox,
                              float* __restrict__ out)
{
    int lane = threadIdx.x;
    float s = 0.0f;
    if (lane < B_SZ) {
        #pragma unroll
        for (int l = 0; l < N_LVL; ++l) {
            int side = 1024 >> l;
            float inv = 1.0f / ((float)side * (float)side);
            s += partial[l * B_SZ + lane] * inv;
        }
        s *= (1.0f / (float)N_LVL);
        if (!hasbox[lane]) s = 0.0f;
    }
    #pragma unroll
    for (int off = 16; off > 0; off >>= 1)
        s += __shfl_xor(s, off, 32);
    if (lane == 0) out[0] = s / (float)B_SZ;
}

extern "C" void kernel_launch(void* const* d_in, const int* in_sizes, int n_in,
                              void* d_out, int out_size, void* d_ws, size_t ws_size,
                              hipStream_t stream)
{
    (void)in_sizes; (void)n_in; (void)out_size;
    const float* atts[N_LVL] = {
        (const float*)d_in[0], (const float*)d_in[1], (const float*)d_in[2],
        (const float*)d_in[3], (const float*)d_in[4]
    };
    const float* bboxs = (const float*)d_in[5];
    const int*   ph    = (const int*)d_in[6];
    const int*   pw    = (const int*)d_in[7];

    uint64_t* boxinfo = (uint64_t*)d_ws;
    float*    partial = (float*)((char*)d_ws + (size_t)N_LVL * B_SZ * N_BOX * sizeof(uint64_t));
    int*      hasbox  = (int*)(partial + N_LVL * B_SZ);

    // Fragment blob offsets (16B-aligned: header is 61824 B = 16 * 3864).
    size_t off = (size_t)N_LVL * B_SZ * N_BOX * sizeof(uint64_t)
               + N_LVL * B_SZ * sizeof(float) + B_SZ * sizeof(int);
    size_t afragOff[N_LVL], bfragOff[N_LVL];
    for (int l = 0; l < N_LVL; ++l) {
        int side = 1024 >> l;
        size_t nt = (size_t)(side >> 4);
        size_t sz = (size_t)B_SZ * nt * 3 * 1024;   // 3 chunks * 32 lanes * 32 B
        afragOff[l] = off; off += sz;
        bfragOff[l] = off; off += sz;
    }
    bool fragPath = (ws_size >= off);

    prep_boxes<<<dim3(N_LVL, B_SZ), N_BOX, 0, stream>>>(bboxs, ph, pw,
                                                        boxinfo, partial, hasbox);

    if (fragPath) {
        for (int l = 0; l < N_LVL; ++l) {
            int side = 1024 >> l;
            build_frags<<<dim3(side >> 4, B_SZ), 32, 0, stream>>>(
                boxinfo,
                (uint8_t*)d_ws + afragOff[l],
                (uint8_t*)d_ws + bfragOff[l],
                side, l);
        }
        for (int l = 0; l < N_LVL; ++l) {
            int side = 1024 >> l;
            int nt   = (side >> 4) * (side >> 4);
            int gx   = (nt + 7) / 8;               // 8 tiles (waves) per block
            level_bce_frag<<<dim3(gx, B_SZ), 256, 0, stream>>>(
                atts[l], side, l,
                (const uint8_t*)d_ws + afragOff[l],
                (const uint8_t*)d_ws + bfragOff[l],
                partial);
        }
    } else {
        for (int l = 0; l < N_LVL; ++l) {
            int side = 1024 >> l;
            int nt   = (side >> 4) * (side >> 4);
            int gx   = (nt + 7) / 8;
            level_bce_inplace<<<dim3(gx, B_SZ), 256, 0, stream>>>(
                atts[l], side, l, boxinfo, partial);
        }
    }

    finalize_loss<<<1, 32, 0, stream>>>(partial, hasbox, (float*)d_out);
}
